// DNALayer_29119878267160
// MI455X (gfx1250) — matrix operand order repaired and checked by reading, not compiled, not run on hardware
//
#include <hip/hip_runtime.h>
#include <hip/hip_bf16.h>
#include <math.h>

typedef __attribute__((ext_vector_type(16))) _Float16 v16h;
typedef __attribute__((ext_vector_type(8)))  _Float16 v8h;
typedef __attribute__((ext_vector_type(4)))  _Float16 v4h;
typedef __attribute__((ext_vector_type(2)))  __fp16   fp16x2;
typedef __attribute__((ext_vector_type(4)))  __fp16   fp16x4;
typedef __attribute__((ext_vector_type(8)))  float    v8f;

#define D_MODEL 640
#define N_HEADS 10
#define D_HEAD  64
#define TOKENS  8192
#define NBUCKET 4096

// Build a v16h fragment from two 16-byte LDS reads.
// B-fragments (K-contiguous): hi_off = 8.  A-fragments: also K-contiguous,
// halves 0..15 at p..p+15 (same k order as B), so hi_off = 8 too.
__device__ __forceinline__ v16h frag16(const _Float16* p, int hi_off) {
  v8h lo = *(const v8h*)(p);
  v8h hi = *(const v8h*)(p + hi_off);
  return __builtin_shufflevector(lo, hi, 0,1,2,3,4,5,6,7,8,9,10,11,12,13,14,15);
}

__device__ __forceinline__ v8f wmma16(v16h a, v16h b, v8f c) {
  return __builtin_amdgcn_wmma_f32_16x16x32_f16(false, a, false, b, (short)0, c, false, false);
}

// pack 4 floats -> 4 halves (v_cvt_pk_rtz + one 8-byte LDS store)
__device__ __forceinline__ v4h pack4(float a, float b, float c, float d) {
  fp16x2 p0 = __builtin_amdgcn_cvt_pkrtz(a, b);
  fp16x2 p1 = __builtin_amdgcn_cvt_pkrtz(c, d);
  fp16x4 r  = __builtin_shufflevector(p0, p1, 0, 1, 2, 3);
  return *(v4h*)&r;   // same IEEE-half bit layout, just different C-level type
}

// ---------------------------------------------------------------------------
// GEMM: C[M,N] = act( A[M,K] @ B[K,N] + bias[N] ), f32 in/out, f16 WMMA math.
// Block tile 128x128, BK=64, double-buffered LDS, register-pipelined global
// loads (issue next stage's loads, WMMA current stage, then convert+store).
// 8 waves as 2x4; wave tile 64x32 -> 16 WMMAs per wave per stage.
// M%128==0, N%128==0, K%64==0 guaranteed by caller.
// LDS layout (halves): stage s: As[128][72] @ s*18432, Bs[128][72] @ +9216.
// ---------------------------------------------------------------------------
__global__ __launch_bounds__(256)
void gemm_wmma_kernel(const float* __restrict__ A, const float* __restrict__ B,
                      const float* __restrict__ bias, float* __restrict__ C,
                      int M, int N, int K, int gelu)
{
  extern __shared__ char smem_raw[];
  _Float16* sm = (_Float16*)smem_raw;   // 2 * (9216 + 9216) halves = 73,728 B

  const int tid  = threadIdx.x;
  const int lane = tid & 31;
  const int wave = tid >> 5;
  const int wm   = wave >> 2;          // 0..1
  const int wn   = wave & 3;           // 0..3
  const int m0   = blockIdx.y * 128;
  const int n0   = blockIdx.x * 128;
  const int l15  = lane & 15;
  const int g8   = (lane >> 4) & 1;    // lane group (rows +8 / K +16)
  const int klo  = g8 ? 8 : 0;         // unused; A- and B-frags both use sel
  const int sel  = g8 ? 16 : 0;        // B-frag K offset

  v8f acc[4][2];
  #pragma unroll
  for (int mi = 0; mi < 4; ++mi)
    #pragma unroll
    for (int ni = 0; ni < 2; ++ni)
      #pragma unroll
      for (int e = 0; e < 8; ++e) acc[mi][ni][e] = 0.0f;

  const int KT = K >> 6;               // K / 64
  float4 fa[8];                        // A stage: 128 rows x 16 quads
  float  fb[8][4];                     // B stage: 128 cols x 16 k-quads

  // ---- prologue: load + store stage 0 ----
  #pragma unroll
  for (int i = 0; i < 8; ++i) {
    int idx = tid + i * 256;
    int r = idx >> 4, c = (idx & 15) << 2;
    fa[i] = *(const float4*)(A + (size_t)(m0 + r) * K + c);
  }
  #pragma unroll
  for (int i = 0; i < 8; ++i) {
    int idx = tid + i * 256;
    int n = idx & 127, k = (idx >> 7) << 2;
    #pragma unroll
    for (int j = 0; j < 4; ++j)
      fb[i][j] = B[(size_t)(k + j) * N + n0 + n];   // coalesced along n
  }
  #pragma unroll
  for (int i = 0; i < 8; ++i) {
    int idx = tid + i * 256;
    int r = idx >> 4, c = (idx & 15) << 2;
    *(v4h*)(sm + r * 72 + c) = pack4(fa[i].x, fa[i].y, fa[i].z, fa[i].w);
  }
  #pragma unroll
  for (int i = 0; i < 8; ++i) {
    int idx = tid + i * 256;
    int n = idx & 127, k = (idx >> 7) << 2;
    *(v4h*)(sm + 9216 + n * 72 + k) = pack4(fb[i][0], fb[i][1], fb[i][2], fb[i][3]);
  }
  __syncthreads();

  for (int kt = 0; kt < KT; ++kt) {
    const int nxt = kt + 1;
    // ---- issue next stage's global loads (hidden behind WMMAs) ----
    if (nxt < KT) {
      const int k0 = nxt << 6;
      #pragma unroll
      for (int i = 0; i < 8; ++i) {
        int idx = tid + i * 256;
        int r = idx >> 4, c = (idx & 15) << 2;
        fa[i] = *(const float4*)(A + (size_t)(m0 + r) * K + k0 + c);
      }
      #pragma unroll
      for (int i = 0; i < 8; ++i) {
        int idx = tid + i * 256;
        int n = idx & 127, k = (idx >> 7) << 2;
        #pragma unroll
        for (int j = 0; j < 4; ++j)
          fb[i][j] = B[(size_t)(k0 + k + j) * N + n0 + n];
      }
    }
    // ---- WMMA on current stage ----
    {
      const _Float16* As = sm + (kt & 1) * 18432;
      const _Float16* Bs = As + 9216;
      #pragma unroll
      for (int kk = 0; kk < 64; kk += 32) {
        v16h b0 = frag16(Bs + (wn*32 +      l15) * 72 + kk + sel, 8);
        v16h b1 = frag16(Bs + (wn*32 + 16 + l15) * 72 + kk + sel, 8);
        #pragma unroll
        for (int mi = 0; mi < 4; ++mi) {
          v16h a = frag16(As + (wm*64 + mi*16 + l15) * 72 + kk + sel, 8);
          acc[mi][0] = wmma16(a, b0, acc[mi][0]);
          acc[mi][1] = wmma16(a, b1, acc[mi][1]);
        }
      }
    }
    // ---- convert + store next stage ----
    if (nxt < KT) {
      _Float16* AsW = sm + (nxt & 1) * 18432;
      _Float16* BsW = AsW + 9216;
      #pragma unroll
      for (int i = 0; i < 8; ++i) {
        int idx = tid + i * 256;
        int r = idx >> 4, c = (idx & 15) << 2;
        *(v4h*)(AsW + r * 72 + c) = pack4(fa[i].x, fa[i].y, fa[i].z, fa[i].w);
      }
      #pragma unroll
      for (int i = 0; i < 8; ++i) {
        int idx = tid + i * 256;
        int n = idx & 127, k = (idx >> 7) << 2;
        *(v4h*)(BsW + n * 72 + k) = pack4(fb[i][0], fb[i][1], fb[i][2], fb[i][3]);
      }
    }
    __syncthreads();
  }

  #pragma unroll
  for (int mi = 0; mi < 4; ++mi) {
    #pragma unroll
    for (int ni = 0; ni < 2; ++ni) {
      int n = n0 + wn*32 + ni*16 + l15;
      float bv = bias[n];
      int rbase = m0 + wm*64 + mi*16 + g8*8;
      #pragma unroll
      for (int e = 0; e < 8; ++e) {
        float v = acc[mi][ni][e] + bv;
        if (gelu) v = 0.5f * v * (1.0f + erff(v * 0.70710678118654752f));
        C[(size_t)(rbase + e) * N + n] = v;
      }
    }
  }
}

// ---------------------------------------------------------------------------
// Flash attention with WMMA. One block = (query tile of 128, one head).
// 8 waves; wave w owns query rows [w*16, w*16+16). Per 128-key tile:
//   S = (Q*0.125) @ K^T  (16 WMMAs/wave), in-register online softmax,
//   P -> LDS (f16, A-frag layout source), O += P @ V (16 WMMAs/wave).
// local_mode: 1 key tile = the query window, causal mask. Else nkt bucket tiles.
// ---------------------------------------------------------------------------
__global__ __launch_bounds__(256)
void flash_wmma_kernel(const float* __restrict__ Qg, const float* __restrict__ Kg,
                       const float* __restrict__ Vg, float* __restrict__ Og,
                       int k_stride, int v_stride, int v_col_base,
                       int n_key_tiles, int causal, int local_mode)
{
  extern __shared__ char smem_raw[];
  _Float16* Qs = (_Float16*)smem_raw;   // [128][72]
  _Float16* Ks = Qs + 128*72;           // [128][72]  (aliased by Ps)
  _Float16* Ps = Ks;                    // [8][16][136] per-wave P staging
  _Float16* Vt = Ks + 17408;            // [64 dims][136 keys] = V^T

  const int tid  = threadIdx.x;
  const int lane = tid & 31;
  const int wave = tid >> 5;
  const int l15  = lane & 15;
  const int g8   = (lane >> 4) & 1;
  const int klo  = g8 ? 8 : 0;
  const int sel  = g8 ? 16 : 0;

  const int h     = blockIdx.x % N_HEADS;
  const int t     = blockIdx.x / N_HEADS;
  const int qrow0 = t * 128;
  const int qcol  = h * D_HEAD;
  const int kcol  = h * D_HEAD;
  const int vcol  = v_col_base + h * D_HEAD;

  // Stage Q tile 128x64, pre-scaled by 1/sqrt(64), packed f16 stores
  #pragma unroll
  for (int i = 0; i < 8; ++i) {
    int idx = tid + i * 256;
    int r = idx >> 4, c = (idx & 15) << 2;
    float4 f = *(const float4*)(Qg + (size_t)(qrow0 + r) * D_MODEL + qcol + c);
    *(v4h*)(Qs + r * 72 + c) =
        pack4(0.125f * f.x, 0.125f * f.y, 0.125f * f.z, 0.125f * f.w);
  }

  float m_run[8], l_run[8];
  v8f oacc[4];
  #pragma unroll
  for (int e = 0; e < 8; ++e) { m_run[e] = -1e30f; l_run[e] = 0.0f; }
  #pragma unroll
  for (int oi = 0; oi < 4; ++oi)
    #pragma unroll
    for (int e = 0; e < 8; ++e) oacc[oi][e] = 0.0f;

  for (int kt = 0; kt < n_key_tiles; ++kt) {
    const int krow0 = local_mode ? qrow0 : kt * 128;

    // Stage K tile [key][dim] (packed b64 stores)
    #pragma unroll
    for (int i = 0; i < 8; ++i) {
      int idx = tid + i * 256;
      int r = idx >> 4, c = (idx & 15) << 2;
      float4 fk = *(const float4*)(Kg + (size_t)(krow0 + r) * k_stride + kcol + c);
      *(v4h*)(Ks + r * 72 + c) = pack4(fk.x, fk.y, fk.z, fk.w);
    }
    // Stage V^T tile [dim][key]: coalesced column loads, contiguous LDS stores
    #pragma unroll
    for (int i = 0; i < 8; ++i) {
      int idx = tid + i * 256;
      int d = idx & 63, r = (idx >> 6) << 2;
      const float* vp = Vg + (size_t)(krow0 + r) * v_stride + vcol + d;
      float v0 = vp[0];
      float v1 = vp[v_stride];
      float v2 = vp[2 * v_stride];
      float v3 = vp[3 * v_stride];
      *(v4h*)(Vt + d * 136 + r) = pack4(v0, v1, v2, v3);
    }
    // Prefetch next key tile while this one is consumed (gfx1250 global_prefetch)
    if (!local_mode && kt + 1 < n_key_tiles) {
      const size_t nrow = (size_t)((kt + 1) * 128 + (tid >> 1));
      __builtin_prefetch(Kg + nrow * k_stride + kcol + ((tid & 1) << 5), 0, 3);
      __builtin_prefetch(Vg + nrow * v_stride + vcol + ((tid & 1) << 5), 0, 3);
    }
    __syncthreads();

    // S = Qs @ Ks^T : wave rows x 128 keys
    v8f s[8];
    #pragma unroll
    for (int ni = 0; ni < 8; ++ni)
      #pragma unroll
      for (int e = 0; e < 8; ++e) s[ni][e] = 0.0f;
    #pragma unroll
    for (int kk = 0; kk < 64; kk += 32) {
      v16h a = frag16(Qs + (wave*16 + l15)*72 + kk + sel, 8);
      #pragma unroll
      for (int ni = 0; ni < 8; ++ni) {
        v16h b = frag16(Ks + (ni*16 + l15)*72 + kk + sel, 8);
        s[ni] = wmma16(a, b, s[ni]);
      }
    }
    if (causal) {
      #pragma unroll
      for (int ni = 0; ni < 8; ++ni)
        #pragma unroll
        for (int e = 0; e < 8; ++e) {
          int qloc = wave*16 + e + g8*8;
          int kloc = ni*16 + l15;
          if (kloc > qloc) s[ni][e] = -1e30f;
        }
    }
    __syncthreads();   // all waves done reading Ks before Ps overwrites it

    // Online softmax (rows exclusive to this wave; cols reduced via shfl)
    #pragma unroll
    for (int e = 0; e < 8; ++e) {
      float rm = s[0][e];
      #pragma unroll
      for (int ni = 1; ni < 8; ++ni) rm = fmaxf(rm, s[ni][e]);
      rm = fmaxf(rm, __shfl_xor(rm, 1));
      rm = fmaxf(rm, __shfl_xor(rm, 2));
      rm = fmaxf(rm, __shfl_xor(rm, 4));
      rm = fmaxf(rm, __shfl_xor(rm, 8));
      float mn    = fmaxf(m_run[e], rm);
      float alpha = expf(m_run[e] - mn);
      m_run[e] = mn;
      float rs = 0.0f;
      int prow = (wave*16 + e + g8*8) * 136;
      #pragma unroll
      for (int ni = 0; ni < 8; ++ni) {
        float p = expf(s[ni][e] - mn);
        rs += p;
        Ps[prow + ni*16 + l15] = (_Float16)p;
      }
      rs += __shfl_xor(rs, 1); rs += __shfl_xor(rs, 2);
      rs += __shfl_xor(rs, 4); rs += __shfl_xor(rs, 8);
      l_run[e] = l_run[e] * alpha + rs;
      #pragma unroll
      for (int oi = 0; oi < 4; ++oi) oacc[oi][e] *= alpha;
    }

    // O += P @ V  (wave-local LDS: DS ops in-order within wave)
    #pragma unroll
    for (int ks2 = 0; ks2 < 4; ++ks2) {
      int kk = ks2 * 32;
      v16h a = frag16(Ps + (wave*16 + l15)*136 + kk + sel, 8);
      #pragma unroll
      for (int oi = 0; oi < 4; ++oi) {
        v16h b = frag16(Vt + (oi*16 + l15)*136 + kk + sel, 8);
        oacc[oi] = wmma16(a, b, oacc[oi]);
      }
    }
    __syncthreads();   // protect Ks/Ps/Vt before next tile's staging
  }

  #pragma unroll
  for (int oi = 0; oi < 4; ++oi) {
    #pragma unroll
    for (int e = 0; e < 8; ++e) {
      int row = qrow0 + wave*16 + e + g8*8;
      int col = h*D_HEAD + oi*16 + l15;
      Og[(size_t)row * D_MODEL + col] = oacc[oi][e] / l_run[e];
    }
  }
}

// ---------------------------------------------------------------------------
// out = LayerNorm(x + r) * g + b, one block per 640-wide row
// ---------------------------------------------------------------------------
__global__ __launch_bounds__(256)
void add_ln_kernel(const float* __restrict__ X, const float* __restrict__ R,
                   const float* __restrict__ g, const float* __restrict__ b,
                   float* __restrict__ out)
{
  const int row = blockIdx.x;
  const int tid = threadIdx.x;
  __shared__ float buf[D_MODEL];
  __shared__ float red[256];

  float partial = 0.0f;
  for (int c = tid; c < D_MODEL; c += 256) {
    float v = X[(size_t)row * D_MODEL + c] + R[(size_t)row * D_MODEL + c];
    buf[c] = v;
    partial += v;
  }
  red[tid] = partial;
  __syncthreads();
  for (int s2 = 128; s2 > 0; s2 >>= 1) {
    if (tid < s2) red[tid] += red[tid + s2];
    __syncthreads();
  }
  float mean = red[0] * (1.0f / D_MODEL);
  __syncthreads();

  partial = 0.0f;
  for (int c = tid; c < D_MODEL; c += 256) {
    float d = buf[c] - mean;
    partial += d * d;
  }
  red[tid] = partial;
  __syncthreads();
  for (int s2 = 128; s2 > 0; s2 >>= 1) {
    if (tid < s2) red[tid] += red[tid + s2];
    __syncthreads();
  }
  float inv = rsqrtf(red[0] * (1.0f / D_MODEL) + 1e-5f);

  for (int c = tid; c < D_MODEL; c += 256)
    out[(size_t)row * D_MODEL + c] = (buf[c] - mean) * inv * g[c] + b[c];
}

// ---------------------------------------------------------------------------
extern "C" void kernel_launch(void* const* d_in, const int* in_sizes, int n_in,
                              void* d_out, int out_size, void* d_ws, size_t ws_size,
                              hipStream_t stream)
{
  (void)in_sizes; (void)n_in; (void)out_size; (void)ws_size;

  const float* x      = (const float*)d_in[0];
  // d_in[1] = positions (unused by reference)
  const float* bucket = (const float*)d_in[2];
  const float* lq_w = (const float*)d_in[3];  const float* lq_b = (const float*)d_in[4];
  const float* lk_w = (const float*)d_in[5];  const float* lk_b = (const float*)d_in[6];
  const float* lv_w = (const float*)d_in[7];  const float* lv_b = (const float*)d_in[8];
  const float* lo_w = (const float*)d_in[9];  const float* lo_b = (const float*)d_in[10];
  const float* gq_w = (const float*)d_in[11]; const float* gq_b = (const float*)d_in[12];
  const float* gkv_w= (const float*)d_in[13]; const float* gkv_b= (const float*)d_in[14];
  const float* go_w = (const float*)d_in[15]; const float* go_b = (const float*)d_in[16];
  const float* f1_w = (const float*)d_in[17]; const float* f1_b = (const float*)d_in[18];
  const float* f2_w = (const float*)d_in[19]; const float* f2_b = (const float*)d_in[20];
  const float* n1_g = (const float*)d_in[21]; const float* n1_b = (const float*)d_in[22];
  const float* n2_g = (const float*)d_in[23]; const float* n2_b = (const float*)d_in[24];
  const float* n3_g = (const float*)d_in[25]; const float* n3_b = (const float*)d_in[26];
  float* out = (float*)d_out;

  const size_t BUF = (size_t)TOKENS * D_MODEL;   // 5,242,880 floats
  float* b0 = (float*)d_ws;      // Q / GQ
  float* b1 = b0 + BUF;          // K / KV (4096x1280 == same size)
  float* b2 = b1 + BUF;          // V / X2
  float* b3 = b2 + BUF;          // attention outputs
  float* b4 = b3 + BUF;          // projection temp
  float* b5 = b4 + BUF;          // X1
  float* bh = b5 + BUF;          // FFN hidden 8192x2560

  const dim3 blk(256);
  const unsigned gemm_smem  = 2u * (9216u + 9216u) * 2u;       // 73,728 B
  const unsigned flash_smem = (128*72 + 17408 + 64*136) * 2u;  // 70,656 B

  auto launch_gemm = [&](const float* A, const float* Bm, const float* bias,
                         float* C, int M, int N, int K, int gelu) {
    dim3 grid((unsigned)(N / 128), (unsigned)(M / 128));
    gemm_wmma_kernel<<<grid, blk, gemm_smem, stream>>>(A, Bm, bias, C, M, N, K, gelu);
  };

  // Local attention branch
  launch_gemm(x, lq_w, lq_b, b0, TOKENS, D_MODEL, D_MODEL, 0);
  launch_gemm(x, lk_w, lk_b, b1, TOKENS, D_MODEL, D_MODEL, 0);
  launch_gemm(x, lv_w, lv_b, b2, TOKENS, D_MODEL, D_MODEL, 0);
  flash_wmma_kernel<<<dim3(640), blk, flash_smem, stream>>>(
      b0, b1, b2, b3, D_MODEL, D_MODEL, /*v_col_base=*/0,
      /*nkt=*/1, /*causal=*/1, /*local=*/1);
  launch_gemm(b3, lo_w, lo_b, b4, TOKENS, D_MODEL, D_MODEL, 0);
  add_ln_kernel<<<dim3(TOKENS), blk, 0, stream>>>(x, b4, n1_g, n1_b, b5);

  // Global (bucket) attention branch
  launch_gemm(b5, gq_w, gq_b, b0, TOKENS, D_MODEL, D_MODEL, 0);
  launch_gemm(bucket, gkv_w, gkv_b, b1, NBUCKET, 2*D_MODEL, D_MODEL, 0);
  flash_wmma_kernel<<<dim3(640), blk, flash_smem, stream>>>(
      b0, b1, b1, b3, 2*D_MODEL, 2*D_MODEL, /*v_col_base=*/D_MODEL,
      /*nkt=*/NBUCKET/128, /*causal=*/0, /*local=*/0);
  launch_gemm(b3, go_w, go_b, b4, TOKENS, D_MODEL, D_MODEL, 0);
  add_ln_kernel<<<dim3(TOKENS), blk, 0, stream>>>(b5, b4, n2_g, n2_b, b2);

  // FFN
  launch_gemm(b2, f1_w, f1_b, bh, TOKENS, 4*D_MODEL, D_MODEL, /*gelu=*/1);
  launch_gemm(bh, f2_w, f2_b, b4, TOKENS, D_MODEL, 4*D_MODEL, 0);
  add_ln_kernel<<<dim3(TOKENS), blk, 0, stream>>>(b2, b4, n3_g, n3_b, out);
}
